// MaskAttentionHead_65506841199208
// MI455X (gfx1250) — matrix operand order repaired and checked by reading, not compiled, not run on hardware
//
#include <hip/hip_runtime.h>
#include <hip/hip_bf16.h>

// ---------------------------------------------------------------------------
// MaskAttentionHead for MI455X (gfx1250, wave32, WMMA bf16 16x16x32)
//   B=4, T=4096, C=1024, H=64
//   q,k,v = x @ W{q,k,v}^T ;  out = softmax(causal(q k^T / 8)) v
// Flash-attention style + async global->LDS double-buffered staging +
// ping-pong software-pipelined B-fragment loads (overlap ds_load with WMMA).
// ---------------------------------------------------------------------------

#define T_SEQ  4096
#define N_BATCH 4
#define C_EMB  1024
#define H_DIM  64

typedef __attribute__((ext_vector_type(16))) __bf16 bf16x16;
typedef __attribute__((ext_vector_type(8)))  float  f32x8;

union AFrag {
  bf16x16 v;
  unsigned short u[16];
  uint4 q[2];
};

__device__ __forceinline__ unsigned short f2bf(float f) {
  __bf16 h = (__bf16)f;                       // hardware cvt (v_cvt_pk_bf16_f32)
  return __builtin_bit_cast(unsigned short, h);
}

__device__ __forceinline__ f32x8 zero8() {
  f32x8 z = {0.f,0.f,0.f,0.f,0.f,0.f,0.f,0.f};
  return z;
}

__device__ __forceinline__ f32x8 wmma_bf16(const AFrag& a, const AFrag& b, f32x8 c) {
  return __builtin_amdgcn_wmma_f32_16x16x32_bf16(
      false, a.v, false, b.v, (short)0, c, false, false);
}

// ---- CDNA5 async global->LDS (ASYNCcnt) -----------------------------------
__device__ __forceinline__ void async_b128(unsigned lds_byte_off, const void* gptr) {
  asm volatile("global_load_async_to_lds_b128 %0, %1, off"
               :: "v"(lds_byte_off), "v"((unsigned long long)(uintptr_t)gptr)
               : "memory");
}
#define WAIT_ASYNC(n) asm volatile("s_wait_asynccnt %0" :: "i"(n) : "memory")

__device__ __forceinline__ unsigned lds_off(const void* p) {
  // flat LDS address: low 32 bits are the workgroup-relative LDS byte offset
  return (unsigned)(unsigned long long)(uintptr_t)p;
}

// ---------------------------------------------------------------------------
// Kernel 1: convert Wq/Wk/Wv (fp32 [64,1024]) to bf16, concatenated in ws.
// ---------------------------------------------------------------------------
__global__ void __launch_bounds__(256)
convert_w_kernel(const float* __restrict__ Wq, const float* __restrict__ Wk,
                 const float* __restrict__ Wv, unsigned short* __restrict__ wb) {
  int i = blockIdx.x * 256 + threadIdx.x;
  if (i < H_DIM * C_EMB) {
    wb[i]                     = f2bf(Wq[i]);
    wb[H_DIM * C_EMB + i]     = f2bf(Wk[i]);
    wb[2 * H_DIM * C_EMB + i] = f2bf(Wv[i]);
  }
}

// ---------------------------------------------------------------------------
// Kernel 2: projections. One workgroup = 64 rows of x, 4 waves x 16 rows.
// W chunks (3 x 64 x 64 bf16 = 24 KB) async-staged into LDS, 2-deep pipeline;
// B-fragments ping-pong pipelined from LDS. 12 WMMAs per 32-K step.
// ---------------------------------------------------------------------------
__global__ void __launch_bounds__(128)
proj_qkv_kernel(const float* __restrict__ x, const unsigned short* __restrict__ wb,
                unsigned short* __restrict__ qb, unsigned short* __restrict__ kbuf,
                unsigned short* __restrict__ vt) {
  __shared__ alignas(16) unsigned short wtile[2][192 * 64];   // 48 KB double buffer

  const int tid  = threadIdx.x;
  const int lane = tid & 31;
  const int wave = tid >> 5;
  const int lr   = lane & 15;
  const int hx   = lane >> 4;
  const int rowbase = blockIdx.x * 64 + wave * 16;
  const int row     = rowbase + lr;

  // async prefetch of one 64-wide W chunk: 384 half-rows of 64B, 12 loads/thread
  auto w_prefetch = [&](int kk, unsigned dst_off) {
    #pragma unroll
    for (int i2 = 0; i2 < 3; ++i2) {
      int h    = tid + 128 * i2;        // 0..383
      int rowm = h >> 1;                // m*64 + n  (0..191)
      int seg  = (h & 1) * 32;          // element offset within 64-wide chunk
      const unsigned short* g = wb + (size_t)rowm * C_EMB + kk + seg;
      unsigned d = dst_off + (unsigned)(rowm * 64 + seg) * 2;
      #pragma unroll
      for (int i = 0; i < 4; ++i)
        async_b128(d + i * 16, (const char*)g + i * 16);
    }
  };

  f32x8 acc[3][4];
  #pragma unroll
  for (int m = 0; m < 3; ++m)
    #pragma unroll
    for (int t = 0; t < 4; ++t) acc[m][t] = zero8();

  const float* xr = x + (size_t)row * C_EMB;
  const unsigned wt0 = lds_off(&wtile[0][0]);
  const unsigned wt1 = lds_off(&wtile[1][0]);

  w_prefetch(0, wt0);

  for (int c = 0; c < C_EMB / 64; ++c) {
    const int kk = c * 64;

    // issue this chunk's x loads early (latency hides under async wait+barrier)
    float4 xv[8];
    {
      const float* xp = xr + kk + 8 * hx;
      #pragma unroll
      for (int i = 0; i < 2; ++i) {
        xv[i*4+0] = *(const float4*)(xp + i*32);
        xv[i*4+1] = *(const float4*)(xp + i*32 + 4);
        xv[i*4+2] = *(const float4*)(xp + i*32 + 16);
        xv[i*4+3] = *(const float4*)(xp + i*32 + 20);
      }
    }

    if (c + 1 < C_EMB / 64) {
      w_prefetch(kk + 64, ((c + 1) & 1) ? wt1 : wt0);
      WAIT_ASYNC(12);                 // in-order: current chunk has landed
    } else {
      WAIT_ASYNC(0);
    }
    __syncthreads();                  // publish LDS tile across waves

    const int buf = c & 1;
    #pragma unroll
    for (int k2 = 0; k2 < 2; ++k2) {  // two 32-wide WMMA K-steps per chunk
      AFrag a;
      {
        float4 v0 = xv[k2*4+0], v1 = xv[k2*4+1], v2 = xv[k2*4+2], v3 = xv[k2*4+3];
        a.u[0]=f2bf(v0.x);  a.u[1]=f2bf(v0.y);  a.u[2]=f2bf(v0.z);  a.u[3]=f2bf(v0.w);
        a.u[4]=f2bf(v1.x);  a.u[5]=f2bf(v1.y);  a.u[6]=f2bf(v1.z);  a.u[7]=f2bf(v1.w);
        a.u[8]=f2bf(v2.x);  a.u[9]=f2bf(v2.y);  a.u[10]=f2bf(v2.z); a.u[11]=f2bf(v2.w);
        a.u[12]=f2bf(v3.x); a.u[13]=f2bf(v3.y); a.u[14]=f2bf(v3.z); a.u[15]=f2bf(v3.w);
      }

      // 12-tile WMMA chain, B-fragments ping-pong pipelined one tile ahead
      auto ldb = [&](AFrag& f, int i) {
        int m = i >> 2, t = i & 3;
        const unsigned short* wp =
            &wtile[buf][(m * 64 + t * 16 + lr) * 64 + k2 * 32 + 8 * hx];
        f.q[0] = *(const uint4*)(wp);
        f.q[1] = *(const uint4*)(wp + 16);
      };
      AFrag bb[2];
      ldb(bb[0], 0);
      #pragma unroll
      for (int i = 0; i < 12; ++i) {
        if (i < 11) ldb(bb[(i + 1) & 1], i + 1);
        acc[i >> 2][i & 3] = wmma_bf16(a, bb[i & 1], acc[i >> 2][i & 3]);
      }
    }
    __syncthreads();                  // all waves done reading before overwrite
  }

  // C/D layout: VGPR r holds element (M = r + 8*hx, N = lr)
  #pragma unroll
  for (int t = 0; t < 4; ++t) {
    #pragma unroll
    for (int r = 0; r < 8; ++r) {
      int grow = rowbase + r + 8 * hx;
      int col  = t * 16 + lr;
      qb[(size_t)grow * H_DIM + col]   = f2bf(acc[0][t][r] * 0.125f);  // 1/sqrt(H)
      kbuf[(size_t)grow * H_DIM + col] = f2bf(acc[1][t][r]);
      int bidx = grow >> 12;
      int tloc = grow & (T_SEQ - 1);
      vt[((size_t)(bidx * H_DIM + col)) * T_SEQ + tloc] = f2bf(acc[2][t][r]);
    }
  }
}

// ---------------------------------------------------------------------------
// Kernel 3: flash attention. Grid (T/64, B); 4 waves x 16 query rows.
// K tile (64x64) and V^T tile (64x64) async-staged into LDS, 2-deep pipeline.
// Per 64-key block: 8 WMMA for S, online softmax, P via wave-private LDS tile
// (C-layout -> A-layout), 8 WMMA for O += P V. All B-frags ping-pong pipelined.
// ---------------------------------------------------------------------------
__global__ void __launch_bounds__(128)
flash_attn_kernel(const unsigned short* __restrict__ qb,
                  const unsigned short* __restrict__ kbuf,
                  const unsigned short* __restrict__ vt,
                  float* __restrict__ out) {
  __shared__ alignas(16) unsigned short ktile[2][64 * 64];   // 16 KB
  __shared__ alignas(16) unsigned short vtile[2][64 * 64];   // 16 KB
  __shared__ alignas(16) unsigned short pbuf[4][16 * 64];    // 8 KB, per-wave

  const int tid  = threadIdx.x;
  const int lane = tid & 31;
  const int wave = tid >> 5;
  const int lr   = lane & 15;
  const int hx   = lane >> 4;
  const int jmax = blockIdx.x;
  const int b    = blockIdx.y;
  const int qrow0 = jmax * 64 + wave * 16;

  // async prefetch of K (64 keys x 64 h) and V^T (64 h x 64 keys): 8 loads/thread
  auto kv_prefetch = [&](int j, int buf) {
    int row = tid >> 1;
    int seg = (tid & 1) * 32;
    const unsigned short* kg = kbuf + ((size_t)(b * T_SEQ + j * 64 + row)) * H_DIM + seg;
    const unsigned short* vg = vt   + ((size_t)(b * H_DIM + row)) * T_SEQ + j * 64 + seg;
    unsigned kd = lds_off(&ktile[buf][0]) + (unsigned)(row * 64 + seg) * 2;
    unsigned vd = lds_off(&vtile[buf][0]) + (unsigned)(row * 64 + seg) * 2;
    #pragma unroll
    for (int i = 0; i < 4; ++i) {
      async_b128(kd + i * 16, (const char*)kg + i * 16);
      async_b128(vd + i * 16, (const char*)vg + i * 16);
    }
  };

  // Q A-fragments over H=64 (two 16x32 chunks), already scaled by 1/8
  AFrag qf[2];
  {
    const unsigned short* qp = qb + ((size_t)(b * T_SEQ + qrow0 + lr)) * H_DIM;
    #pragma unroll
    for (int kc = 0; kc < 2; ++kc) {
      qf[kc].q[0] = *(const uint4*)(qp + kc * 32 + 8 * hx);
      qf[kc].q[1] = *(const uint4*)(qp + kc * 32 + 16 + 8 * hx);
    }
  }

  float mrow[8], lrow[8];
  f32x8 o[4];
  #pragma unroll
  for (int r = 0; r < 8; ++r) { mrow[r] = -1e30f; lrow[r] = 0.f; }
  #pragma unroll
  for (int t = 0; t < 4; ++t) o[t] = zero8();

  unsigned short* pb = pbuf[wave];

  kv_prefetch(0, 0);

  for (int j = 0; j <= jmax; ++j) {
    const int buf = j & 1;
    if (j < jmax) {
      kv_prefetch(j + 1, buf ^ 1);
      WAIT_ASYNC(8);                 // in-order: tile j has landed
    } else {
      WAIT_ASYNC(0);
    }
    __syncthreads();                 // publish K/V tiles across waves

    // ---- S = Q K^T : 8-WMMA chain, K-frags pipelined ----
    f32x8 s[4];
    #pragma unroll
    for (int t = 0; t < 4; ++t) s[t] = zero8();
    {
      auto ldk = [&](AFrag& f, int i) {
        int t = i >> 1, kc = i & 1;
        const unsigned short* kp = &ktile[buf][(t * 16 + lr) * 64 + kc * 32 + 8 * hx];
        f.q[0] = *(const uint4*)(kp);
        f.q[1] = *(const uint4*)(kp + 16);
      };
      AFrag bb[2];
      ldk(bb[0], 0);
      #pragma unroll
      for (int i = 0; i < 8; ++i) {
        if (i < 7) ldk(bb[(i + 1) & 1], i + 1);
        s[i >> 1] = wmma_bf16(qf[i & 1], bb[i & 1], s[i >> 1]);
      }
    }

    // ---- causal mask on the diagonal block ----
    if (j == jmax) {
      #pragma unroll
      for (int t = 0; t < 4; ++t)
        #pragma unroll
        for (int r = 0; r < 8; ++r) {
          int rrow = qrow0 + r + 8 * hx;
          int ccol = j * 64 + t * 16 + lr;
          if (ccol > rrow) s[t][r] = -1e30f;
        }
    }

    // ---- online softmax; each row lives in a 16-lane half ----
    #pragma unroll
    for (int r = 0; r < 8; ++r) {
      float mx = fmaxf(fmaxf(s[0][r], s[1][r]), fmaxf(s[2][r], s[3][r]));
      #pragma unroll
      for (int d = 1; d <= 8; d <<= 1) mx = fmaxf(mx, __shfl_xor(mx, d, 32));
      float mn = fmaxf(mrow[r], mx);
      float al = __expf(mrow[r] - mn);
      mrow[r] = mn;
      float rs = 0.f;
      #pragma unroll
      for (int t = 0; t < 4; ++t) {
        float p = __expf(s[t][r] - mn);
        s[t][r] = p;
        rs += p;
        pb[(r + 8 * hx) * 64 + t * 16 + lr] = f2bf(p);
      }
      #pragma unroll
      for (int d = 1; d <= 8; d <<= 1) rs += __shfl_xor(rs, d, 32);
      lrow[r] = lrow[r] * al + rs;
      #pragma unroll
      for (int t = 0; t < 4; ++t) o[t][r] *= al;
    }

    asm volatile("s_wait_dscnt 0x0" ::: "memory");   // wave-private P RAW

    // ---- P back in A-layout ----
    AFrag pf[2];
    #pragma unroll
    for (int kc = 0; kc < 2; ++kc) {
      const unsigned short* pp = pb + lr * 64 + kc * 32 + 8 * hx;
      pf[kc].q[0] = *(const uint4*)(pp);
      pf[kc].q[1] = *(const uint4*)(pp + 16);
    }

    // ---- O += P V : 8-WMMA chain, V-frags pipelined ----
    {
      auto ldv = [&](AFrag& f, int i) {
        int t = i >> 1, kc = i & 1;
        const unsigned short* vp = &vtile[buf][(t * 16 + lr) * 64 + kc * 32 + 8 * hx];
        f.q[0] = *(const uint4*)(vp);
        f.q[1] = *(const uint4*)(vp + 16);
      };
      AFrag bb[2];
      ldv(bb[0], 0);
      #pragma unroll
      for (int i = 0; i < 8; ++i) {
        if (i < 7) ldv(bb[(i + 1) & 1], i + 1);
        o[i >> 1] = wmma_bf16(pf[i & 1], bb[i & 1], o[i >> 1]);
      }
    }
    __syncthreads();                 // all waves done with tile j before reuse
  }

  // ---- epilogue: O / l, fp32 out [B,T,64] ----
  #pragma unroll
  for (int r = 0; r < 8; ++r) {
    float inv = 1.0f / lrow[r];
    int rrow = qrow0 + r + 8 * hx;
    #pragma unroll
    for (int t = 0; t < 4; ++t)
      out[((size_t)(b * T_SEQ + rrow)) * H_DIM + t * 16 + lr] = o[t][r] * inv;
  }
}

// ---------------------------------------------------------------------------
// Workspace layout (bytes):
//   [0,      384K)  wb : 3 x 64 x 1024 bf16 weights
//   [+384K,  +2MB)  qb : [B*T,64] bf16 (scaled)
//   [..,     +2MB)  kb : [B*T,64] bf16
//   [..,     +2MB)  vt : [B,64,T] bf16 (transposed)
// Total ~6.4 MB.
// ---------------------------------------------------------------------------
extern "C" void kernel_launch(void* const* d_in, const int* in_sizes, int n_in,
                              void* d_out, int out_size, void* d_ws, size_t ws_size,
                              hipStream_t stream) {
  const float* x  = (const float*)d_in[0];
  const float* Wq = (const float*)d_in[1];
  const float* Wk = (const float*)d_in[2];
  const float* Wv = (const float*)d_in[3];
  float* out = (float*)d_out;

  char* ws = (char*)d_ws;
  const size_t wb_bytes = (size_t)3 * H_DIM * C_EMB * 2;
  const size_t qk_bytes = (size_t)N_BATCH * T_SEQ * H_DIM * 2;
  unsigned short* wb = (unsigned short*)ws;
  unsigned short* qb = (unsigned short*)(ws + wb_bytes);
  unsigned short* kb = (unsigned short*)(ws + wb_bytes + qk_bytes);
  unsigned short* vt = (unsigned short*)(ws + wb_bytes + 2 * qk_bytes);

  convert_w_kernel<<<(H_DIM * C_EMB + 255) / 256, 256, 0, stream>>>(Wq, Wk, Wv, wb);

  proj_qkv_kernel<<<(N_BATCH * T_SEQ) / 64, 128, 0, stream>>>(x, wb, qb, kb, vt);

  dim3 grid(T_SEQ / 64, N_BATCH);
  flash_attn_kernel<<<grid, 128, 0, stream>>>(qb, kb, vt, out);
}
